// AttentionBlock_73323681677485
// MI455X (gfx1250) — matrix operand order repaired and checked by reading, not compiled, not run on hardware
//
#include <hip/hip_runtime.h>

typedef _Float16 v16h __attribute__((ext_vector_type(16)));
typedef _Float16 v2h  __attribute__((ext_vector_type(2)));
typedef float    v8f  __attribute__((ext_vector_type(8)));

#define B_   4
#define T_   1024
#define D_   64
#define XH_STRIDE 68    /* halves; 34-dword row stride -> 16 distinct banks for column reads */
#define S_STRIDE  1025  /* floats; bank = (row + col) % 64 -> conflict-free rows AND columns */

#define XH_BYTES  (T_ * XH_STRIDE * 2)   /* 139264 */
#define S_BYTES   (16 * S_STRIDE * 4)    /* 65600  */
#define XQ_BYTES  (16 * D_ * 4)          /* 4096   */
#define SCL_BYTES (D_ * 4)               /* 256    */
#define INV_BYTES (64)
#define SMEM_BYTES (XH_BYTES + S_BYTES + XQ_BYTES + SCL_BYTES + INV_BYTES)

__device__ __forceinline__ float fast_tanh(float v) {
#if __has_builtin(__builtin_amdgcn_tanhf)
    return __builtin_amdgcn_tanhf(v);            // v_tanh_f32 (CDNA5 trans op)
#else
    // tanh(v) = 1 - 2/(exp(2v)+1); exp(2v) = exp2(v * 2*log2(e))
    float e = __builtin_amdgcn_exp2f(v * 2.885390081777927f);  // v_exp_f32
    return 1.0f - 2.0f * __builtin_amdgcn_rcpf(e + 1.0f);      // v_rcp_f32
#endif
}

__global__ __launch_bounds__(256) void attn_block_kernel(
    const float* __restrict__ x, const float* __restrict__ scale,
    const float* __restrict__ gamma, const float* __restrict__ beta,
    const float* __restrict__ mmean, const float* __restrict__ mvar,
    float* __restrict__ out)
{
    extern __shared__ __align__(16) char smem[];
    _Float16* xh  = (_Float16*)smem;                                   // x[b] in f16, padded rows
    float*    S   = (float*)(smem + XH_BYTES);                         // scores -> exp(scores-max)
    float*    xq  = (float*)(smem + XH_BYTES + S_BYTES);               // 16 query rows, f32
    float*    scl = (float*)(smem + XH_BYTES + S_BYTES + XQ_BYTES);    // scale[64]
    float*    inv = (float*)(smem + XH_BYTES + S_BYTES + XQ_BYTES + SCL_BYTES); // 1/rowsum[16]

    const int tid = threadIdx.x;
    const int b   = blockIdx.x >> 6;          // 4 batches
    const int q0  = (blockIdx.x & 63) << 4;   // 64 query tiles of 16
    const float* xb = x + (size_t)b * (T_ * D_);

    // ---- Phase 0: stage x[b] -> LDS f16 (coalesced), query tile -> f32 ----
    {
        const float2* xp = (const float2*)xb;
        for (int i = tid; i < T_ * D_ / 2; i += 256) {
            float2 v = xp[i];
            const int k = i >> 5;             // 32 float2 per 64-wide row
            const int c = (i & 31) << 1;
            v2h h; h.x = (_Float16)v.x; h.y = (_Float16)v.y;
            *(v2h*)(xh + k * XH_STRIDE + c) = h;
        }
        for (int i = tid; i < 16 * D_; i += 256) xq[i] = xb[q0 * D_ + i];
        if (tid < D_) scl[tid] = scale[tid];
    }
    __syncthreads();

    // ---- Phase 1: scores[q][k] = sum_d scale[d]*tanh(xq+xk)  (trans-pipe bound) ----
    {
        const int q  = tid >> 4;
        const int ks = tid & 15;
        const float* xqr = xq + q * D_;
        for (int k = ks; k < T_; k += 16) {
            const _Float16* xkr = xh + k * XH_STRIDE;
            float s0 = 0.f, s1 = 0.f, s2 = 0.f, s3 = 0.f;  // 4 accumulators for ILP
            #pragma unroll
            for (int d = 0; d < D_; d += 4) {
                s0 += scl[d + 0] * fast_tanh(xqr[d + 0] + (float)xkr[d + 0]);
                s1 += scl[d + 1] * fast_tanh(xqr[d + 1] + (float)xkr[d + 1]);
                s2 += scl[d + 2] * fast_tanh(xqr[d + 2] + (float)xkr[d + 2]);
                s3 += scl[d + 3] * fast_tanh(xqr[d + 3] + (float)xkr[d + 3]);
            }
            S[q * S_STRIDE + k] = (s0 + s1) + (s2 + s3);
        }
    }
    __syncthreads();

    // ---- Phase 2: softmax over k (wave handles 2 rows; wave32 shuffles) ----
    {
        const int w = tid >> 5, lane = tid & 31;
        for (int rr = 0; rr < 2; ++rr) {
            const int r = (w << 1) + rr;
            float* Sr = S + r * S_STRIDE;
            float m = -3.0e38f;
            for (int j = lane; j < T_; j += 32) m = fmaxf(m, Sr[j]);
            #pragma unroll
            for (int off = 16; off; off >>= 1) m = fmaxf(m, __shfl_xor(m, off));
            float sum = 0.f;
            for (int j = lane; j < T_; j += 32) {
                float e = __builtin_amdgcn_exp2f((Sr[j] - m) * 1.4426950408889634f);
                Sr[j] = e;
                sum += e;
            }
            #pragma unroll
            for (int off = 16; off; off >>= 1) sum += __shfl_xor(sum, off);
            if (lane == 0) inv[r] = 1.0f / sum;
        }
    }
    __syncthreads();

    // ---- Phase 3: ctx = P @ x via v_wmma_f32_16x16x32_f16; fused BN + residual ----
    const int w = tid >> 5;
    if (w < 4) {                                   // wave w owns d-tile [w*16, w*16+16)
        const int lane = tid & 31;
        const int M    = lane & 15;                // A row / C-D col index base
        const int dcol = (w << 4) | M;             // global d for this lane's column
        const int kgA  = (lane >> 4) << 3;         // A: +8 K-offset for upper half-wave
        const int kgB  = (lane >> 4) << 4;         // unused; B indexes with kgA (+8), same K order as A
        const float rinv = inv[M];
        const float* SA = S + M * S_STRIDE;

        v8f c = {0.f, 0.f, 0.f, 0.f, 0.f, 0.f, 0.f, 0.f};
        for (int kb = 0; kb < T_; kb += 32) {
            v16h a, bm;
            #pragma unroll
            for (int h = 0; h < 16; ++h) {         // A 16x32 f16 layout (ISA 7.12.2)
                const int K = ((h >> 3) << 4) | kgA | (h & 7);
                a[h] = (_Float16)(SA[kb + K] * rinv);
            }
            #pragma unroll
            for (int h = 0; h < 16; ++h)           // B 32x16 f16: same K order as A (lanes 0-15 K=0..7,16..23; 16-31 K=8..15,24..31)
                bm[h] = xh[(kb + (((h >> 3) << 4) | kgA | (h & 7))) * XH_STRIDE + dcol];
            c = __builtin_amdgcn_wmma_f32_16x16x32_f16(
                    false, a, false, bm, (short)0, c, false, false);
        }

        // BN (inference) + residual; d is fixed per lane -> scalar param loads
        const float g  = gamma[dcol];
        const float be = beta[dcol];
        const float mm = mmean[dcol];
        const float rs = rsqrtf(mvar[dcol] + 1.0e-3f);
        const int rbase = q0 + ((lane >> 4) << 3); // C layout: VGPR r -> M = r (+8 upper half)
        float* outb = out + (size_t)b * (T_ * D_);
        #pragma unroll
        for (int r = 0; r < 8; ++r) {
            const int idx = (rbase + r) * D_ + dcol;
            outb[idx] = xb[idx] + g * (c[r] - mm) * rs + be;
        }
    }
}

extern "C" void kernel_launch(void* const* d_in, const int* in_sizes, int n_in,
                              void* d_out, int out_size, void* d_ws, size_t ws_size,
                              hipStream_t stream) {
    (void)in_sizes; (void)n_in; (void)out_size; (void)d_ws; (void)ws_size;
    const float* x     = (const float*)d_in[0];
    const float* scale = (const float*)d_in[1];
    const float* gamma = (const float*)d_in[2];
    const float* beta  = (const float*)d_in[3];
    const float* mmean = (const float*)d_in[4];
    const float* mvar  = (const float*)d_in[5];
    float* out = (float*)d_out;

    // Allow >64KB dynamic LDS (CDNA5 WGP has 320KB); idempotent, capture-safe.
    hipFuncSetAttribute((const void*)attn_block_kernel,
                        hipFuncAttributeMaxDynamicSharedMemorySize, SMEM_BYTES);

    dim3 grid(B_ * (T_ / 16));   // 256 workgroups: (batch, 16-query tile)
    dim3 block(256);             // 8 wave32
    attn_block_kernel<<<grid, block, SMEM_BYTES, stream>>>(
        x, scale, gamma, beta, mmean, mvar, out);
}
